// PPINet_40733469835824
// MI455X (gfx1250) — compile-verified, matrix operand order checked by reading, not run on hardware
//
#include <hip/hip_runtime.h>
#include <hip/hip_bf16.h>

// ---------------------------------------------------------------------------
// Problem constants (from reference)
// ---------------------------------------------------------------------------
#define NN      10000        // nodes
#define E0      160000       // raw edges
#define ETOT    170000       // edges + self loops
#define CDIM    256          // per-head channels

typedef __bf16 bf16;
typedef __attribute__((ext_vector_type(16))) __bf16 v16bf;
typedef __attribute__((ext_vector_type(8)))  __bf16 v8bf;
typedef __attribute__((ext_vector_type(8)))  float  v8f;

union FragAB { v16bf m; v8bf h[2]; };

// LDS byte-offset of a pointer into a __shared__ array (generic -> AS3 -> u32)
template <typename T>
__device__ __forceinline__ unsigned ldsoff(const T* p) {
    return (unsigned)(unsigned long long)
        (const __attribute__((address_space(3))) T*)p;
}

// CDNA5 async copy: LDS[lds] <- 16 bytes at global address g (per lane).
__device__ __forceinline__ void async_ld_b128(unsigned lds, const void* g) {
    asm volatile("global_load_async_to_lds_b128 %0, %1, off"
                 :: "v"(lds), "v"(g) : "memory");
}
// ASYNCcnt completes in order: waiting <=4 guarantees the *previous* 4-load
// tile has landed while the next tile's 4 loads remain in flight.
__device__ __forceinline__ void wait_async_le4() {
    asm volatile("s_wait_asynccnt 0x4" ::: "memory");
}
__device__ __forceinline__ void wait_async_le0() {
    asm volatile("s_wait_asynccnt 0x0" ::: "memory");
}

// ---------------------------------------------------------------------------
// WMMA GEMM:  C[M,Nout] = A[M,K] * B[Nout,K]^T   (NT, bf16 in, f32 out)
// Requirements: K % 32 == 0, Nout % 128 == 0 (true for all call sites).
// 256 threads = 8 waves; block tile 128x128; wave tile 32x64 (2x4 frags).
// Double-buffered LDS; tiles staged with global_load_async_to_lds_b128.
// ---------------------------------------------------------------------------
#define GTM 128
#define GTN 128
#define GTK 32
#define LSTR 40   // bf16 elems per LDS row (32 + 8 pad; 80B rows, 16B aligned)

__global__ __launch_bounds__(256)
void gemm_nt_bf16(const bf16* __restrict__ A, const bf16* __restrict__ B,
                  float* __restrict__ C, int M, int Nout, int K)
{
    __shared__ bf16 sA[2][GTM * LSTR];
    __shared__ bf16 sB[2][GTN * LSTR];

    const int m0   = blockIdx.y * GTM;
    const int n0   = blockIdx.x * GTN;
    const int t    = threadIdx.x;
    const int wave = t >> 5;
    const int lane = t & 31;
    const int wm   = (wave >> 1) * 32;   // 0,32,64,96
    const int wn   = (wave & 1) * 64;    // 0,64
    const int half = lane >> 4;          // 0/1 (K-half selector)
    const int r    = lane & 15;          // row of A-frag / col of B-frag

    v8f acc[2][4];
    #pragma unroll
    for (int i = 0; i < 2; ++i)
        #pragma unroll
        for (int j = 0; j < 4; ++j)
            acc[i][j] = {};

    // loader mapping: thread t owns row (t>>1), 16-col half (t&1); 2x b128 each
    const int lrow = t >> 1;             // 0..127
    const int lcol = (t & 1) * 16;       // 0 or 16
    // branchless M-boundary: clamp row; garbage lands only in discarded C rows
    const int garow = (m0 + lrow < M) ? (m0 + lrow) : (M - 1);
    const int gbrow = n0 + lrow;         // Nout % 128 == 0 -> always valid

    const unsigned ldsA[2] = { ldsoff(&sA[0][lrow * LSTR + lcol]),
                               ldsoff(&sA[1][lrow * LSTR + lcol]) };
    const unsigned ldsB[2] = { ldsoff(&sB[0][lrow * LSTR + lcol]),
                               ldsoff(&sB[1][lrow * LSTR + lcol]) };
    const bf16* gA = A + (size_t)garow * K + lcol;
    const bf16* gB = B + (size_t)gbrow * K + lcol;

    auto issue = [&](int it, int buf) {
        const int k0 = it * GTK;
        async_ld_b128(ldsA[buf],      gA + k0);
        async_ld_b128(ldsA[buf] + 16, gA + k0 + 8);
        async_ld_b128(ldsB[buf],      gB + k0);
        async_ld_b128(ldsB[buf] + 16, gB + k0 + 8);
    };

    const int nIter = K >> 5;            // K % 32 == 0
    issue(0, 0);
    for (int it = 0; it < nIter; ++it) {
        const int cur = it & 1;
        if (it + 1 < nIter) {
            issue(it + 1, cur ^ 1);      // prefetch next tile into other buffer
            wait_async_le4();            // current tile complete (in-order)
        } else {
            wait_async_le0();
        }
        __syncthreads();                 // all waves' tile data visible

        // ---- fragments: lane l<16: K 0..7 & 16..23, lane>=16: K 8..15 & 24..31
        const bf16* bA = sA[cur];
        const bf16* bB = sB[cur];
        FragAB af[2], bfr[4];
        #pragma unroll
        for (int i = 0; i < 2; ++i) {
            const bf16* p = &bA[(wm + i*16 + r) * LSTR + half*8];
            af[i].h[0] = *(const v8bf*)p;
            af[i].h[1] = *(const v8bf*)(p + 16);
        }
        #pragma unroll
        for (int j = 0; j < 4; ++j) {
            const bf16* p = &bB[(wn + j*16 + r) * LSTR + half*8];
            bfr[j].h[0] = *(const v8bf*)p;
            bfr[j].h[1] = *(const v8bf*)(p + 16);
        }
        #pragma unroll
        for (int i = 0; i < 2; ++i)
            #pragma unroll
            for (int j = 0; j < 4; ++j)
                acc[i][j] = __builtin_amdgcn_wmma_f32_16x16x32_bf16(
                    false, af[i].m, false, bfr[j].m, (short)0, acc[i][j],
                    false, false);
        __syncthreads();                 // done reading before buffer reuse
    }

    // ---- store: D layout lane l: N=l%16; VGPR v: M = v + 8*(l/16)
    #pragma unroll
    for (int i = 0; i < 2; ++i)
        #pragma unroll
        for (int j = 0; j < 4; ++j)
            #pragma unroll
            for (int v = 0; v < 8; ++v) {
                int row = m0 + wm + i*16 + half*8 + v;
                int col = n0 + wn + j*16 + r;
                if (row < M) C[(size_t)row * Nout + col] = acc[i][j][v];
            }
}

// ---------------------------------------------------------------------------
// fp32 -> bf16 conversion with optional K padding; 2-D grid, division-free
// ---------------------------------------------------------------------------
__global__ void cvt_bf16_pad(const float* __restrict__ src, bf16* __restrict__ dst,
                             int Kin, int Kpad)
{
    const int r = blockIdx.y;
    const int c = blockIdx.x * blockDim.x + threadIdx.x;
    if (c >= Kpad) return;
    dst[(size_t)r * Kpad + c] = (bf16)(c < Kin ? src[(size_t)r * Kin + c] : 0.0f);
}

// ---------------------------------------------------------------------------
// Helpers
// ---------------------------------------------------------------------------
__device__ __forceinline__ unsigned f2ord(float f) {
    unsigned u = __float_as_uint(f);
    return (u & 0x80000000u) ? ~u : (u | 0x80000000u);
}
__device__ __forceinline__ float ord2f(unsigned u) {
    return (u & 0x80000000u) ? __uint_as_float(u & 0x7fffffffu)
                             : __uint_as_float(~u);
}
__device__ __forceinline__ void edge_sd(int e, const int* __restrict__ ei,
                                        int& s, int& d) {
    if (e < E0) { s = ei[e]; d = ei[E0 + e]; }
    else        { s = d = e - E0; }
}

__global__ void fill_u32(unsigned* p, unsigned v, int n) {
    int i = blockIdx.x * blockDim.x + threadIdx.x;
    if (i < n) p[i] = v;
}
__global__ void fill_f32(float* p, float v, int n) {
    int i = blockIdx.x * blockDim.x + threadIdx.x;
    if (i < n) p[i] = v;
}

// per-(node,head) attention logits: as[n,h] = <h[n,h,:], a_s[h,:]>, same for a_d
template <int H>
__global__ __launch_bounds__(256)
void alphas_kernel(const float* __restrict__ h, const float* __restrict__ aw_s,
                   const float* __restrict__ aw_d, float* __restrict__ outs,
                   float* __restrict__ outd)
{
    __shared__ float reds[256];
    __shared__ float redd[256];
    const int n = blockIdx.x, t = threadIdx.x;
    constexpr int HC = H << 8;
    #pragma unroll
    for (int hh = 0; hh < H; ++hh) {
        float v = h[(size_t)n * HC + (hh << 8) + t];
        reds[t] = v * aw_s[(hh << 8) + t];
        redd[t] = v * aw_d[(hh << 8) + t];
        __syncthreads();
        for (int sdt = 128; sdt > 0; sdt >>= 1) {
            if (t < sdt) { reds[t] += reds[t+sdt]; redd[t] += redd[t+sdt]; }
            __syncthreads();
        }
        if (t == 0) { outs[n*H + hh] = reds[0]; outd[n*H + hh] = redd[0]; }
        __syncthreads();
    }
}

template <int H>
__global__ void edge_max_kernel(const float* __restrict__ as_, const float* __restrict__ ad_,
                                const int* __restrict__ ei, unsigned* __restrict__ emax)
{
    int idx = blockIdx.x * blockDim.x + threadIdx.x;
    if (idx >= ETOT * H) return;
    int e = idx / H, hh = idx - e * H;
    int s, d; edge_sd(e, ei, s, d);
    float v = as_[s*H + hh] + ad_[d*H + hh];
    v = v > 0.0f ? v : 0.2f * v;                 // leaky_relu 0.2
    atomicMax(&emax[d*H + hh], f2ord(v));
}

template <int H>
__global__ void edge_exp_kernel(const float* __restrict__ as_, const float* __restrict__ ad_,
                                const int* __restrict__ ei, const unsigned* __restrict__ emax,
                                float* __restrict__ denom, float* __restrict__ coef)
{
    int idx = blockIdx.x * blockDim.x + threadIdx.x;
    if (idx >= ETOT * H) return;
    int e = idx / H, hh = idx - e * H;
    int s, d; edge_sd(e, ei, s, d);
    float v = as_[s*H + hh] + ad_[d*H + hh];
    v = v > 0.0f ? v : 0.2f * v;
    float ex = __expf(v - ord2f(emax[d*H + hh]));
    coef[idx] = ex;
    atomicAdd(&denom[d*H + hh], ex);
}

template <int H>
__global__ void coef_div_kernel(float* __restrict__ coef, const float* __restrict__ denom,
                                const int* __restrict__ ei)
{
    int idx = blockIdx.x * blockDim.x + threadIdx.x;
    if (idx >= ETOT * H) return;
    int e = idx / H, hh = idx - e * H;
    int s, d; edge_sd(e, ei, s, d);
    coef[idx] = coef[idx] / (denom[d*H + hh] + 1e-16f);
}

// bandwidth-critical scatter: agg[dst,h,c] += hbf[src,h,c] * coef[e,h]
// bf16 gather halves random-read traffic; f32 atomic accumulation.
// H templated: / (H*256) becomes mul-hi/shift, not a runtime divide.
template <int H>
__global__ __launch_bounds__(256)
void aggregate_kernel(const bf16* __restrict__ hbf, const float* __restrict__ coef,
                      const int* __restrict__ ei, float* __restrict__ agg)
{
    constexpr int HC  = H << 8;
    constexpr int tot = ETOT * HC;               // <= 261M, fits int32
    const int str = blockDim.x * gridDim.x;
    for (int idx = blockIdx.x * blockDim.x + threadIdx.x; idx < tot; idx += str) {
        int e  = idx / HC;
        int rr = idx - e * HC;
        int hh = rr >> 8;
        int s, d; edge_sd(e, ei, s, d);
        float cf = coef[e*H + hh];
        float hv = (float)hbf[(size_t)s * HC + rr];
        atomicAdd(&agg[(size_t)d * HC + rr], hv * cf);
    }
}

// 2-D grids: row = blockIdx.y, channel = blockIdx.x*256+tid (HC % 256 == 0)
template <int HC>
__global__ void finalize_cat_kernel(const float* __restrict__ agg, const float* __restrict__ b,
                                    const float* __restrict__ skip, const float* __restrict__ lb,
                                    float* __restrict__ xout)
{
    const int n = blockIdx.y;
    const int c = blockIdx.x * 256 + threadIdx.x;
    const size_t idx = (size_t)n * HC + c;
    float v = agg[idx] + b[c] + skip[idx] + lb[c];
    xout[idx] = v > 0.0f ? v : (__expf(v) - 1.0f);    // ELU
}

template <int H>
__global__ void finalize_mean_kernel(const float* __restrict__ agg, const float* __restrict__ b,
                                     const float* __restrict__ skip, const float* __restrict__ lb,
                                     float* __restrict__ xout)
{
    const int n = blockIdx.y;
    const int c = threadIdx.x;                   // CDIM == 256 == blockDim
    constexpr int HC = H << 8;
    float s = 0.0f;
    #pragma unroll
    for (int hh = 0; hh < H; ++hh) s += agg[(size_t)n * HC + (hh << 8) + c];
    const size_t idx = (size_t)n * CDIM + c;
    xout[idx] = s / (float)H + b[c] + skip[idx] + lb[c];
}

// ---------------------------------------------------------------------------
// Host-side layer driver (all inputs to GEMMs are pre-converted bf16, K%32==0)
// ---------------------------------------------------------------------------
template <int H, bool CONCAT>
static void gat_layer(const bf16* xin, int K,
                      const bf16* Wbf, const bf16* lwbf,
                      const float* aws, const float* awd,
                      const float* b, const float* lb,
                      float* h, bf16* hbf, float* skip, float* agg,
                      float* asb, float* adb, unsigned* emax, float* denom,
                      float* coef, const int* ei, float* xout, hipStream_t stream)
{
    constexpr int Nout     = H * CDIM;
    constexpr int NoutSkip = CONCAT ? Nout : CDIM;

    dim3 gg(Nout / GTN, (NN + GTM - 1) / GTM);
    gemm_nt_bf16<<<gg, 256, 0, stream>>>(xin, Wbf,  h,    NN, Nout,     K);
    dim3 gs(NoutSkip / GTN, (NN + GTM - 1) / GTM);
    gemm_nt_bf16<<<gs, 256, 0, stream>>>(xin, lwbf, skip, NN, NoutSkip, K);

    alphas_kernel<H><<<NN, 256, 0, stream>>>(h, aws, awd, asb, adb);

    const int nh = NN * H;
    fill_u32<<<(nh + 255) / 256, 256, 0, stream>>>(emax, 0x007FFFFFu, nh); // ord(-inf)
    fill_f32<<<(nh + 255) / 256, 256, 0, stream>>>(denom, 0.0f, nh);
    const int na = NN * Nout;
    fill_f32<<<(na + 255) / 256, 256, 0, stream>>>(agg, 0.0f, na);

    const int eh = ETOT * H;
    edge_max_kernel<H><<<(eh + 255) / 256, 256, 0, stream>>>(asb, adb, ei, emax);
    edge_exp_kernel<H><<<(eh + 255) / 256, 256, 0, stream>>>(asb, adb, ei, emax, denom, coef);
    coef_div_kernel<H><<<(eh + 255) / 256, 256, 0, stream>>>(coef, denom, ei);

    cvt_bf16_pad<<<dim3(Nout / 256, NN), 256, 0, stream>>>(h, hbf, Nout, Nout);
    aggregate_kernel<H><<<8192, 256, 0, stream>>>(hbf, coef, ei, agg);

    if (CONCAT)
        finalize_cat_kernel<Nout><<<dim3(Nout / 256, NN), 256, 0, stream>>>(agg, b, skip, lb, xout);
    else
        finalize_mean_kernel<H><<<dim3(1, NN), 256, 0, stream>>>(agg, b, skip, lb, xout);
}

static void cvt(const float* s, bf16* d, int R, int Kin, int Kpad, hipStream_t st) {
    cvt_bf16_pad<<<dim3((Kpad + 255) / 256, R), 256, 0, st>>>(s, d, Kin, Kpad);
}

extern "C" void kernel_launch(void* const* d_in, const int* in_sizes, int n_in,
                              void* d_out, int out_size, void* d_ws, size_t ws_size,
                              hipStream_t stream)
{
    (void)in_sizes; (void)n_in; (void)out_size; (void)ws_size;
    const float* x   = (const float*)d_in[0];
    const int*   ei  = (const int*)  d_in[1];
    const float* W1  = (const float*)d_in[2];
    const float* a1s = (const float*)d_in[3];
    const float* a1d = (const float*)d_in[4];
    const float* b1  = (const float*)d_in[5];
    const float* lw1 = (const float*)d_in[6];
    const float* lb1 = (const float*)d_in[7];
    const float* W2  = (const float*)d_in[8];
    const float* a2s = (const float*)d_in[9];
    const float* a2d = (const float*)d_in[10];
    const float* b2  = (const float*)d_in[11];
    const float* lw2 = (const float*)d_in[12];
    const float* lb2 = (const float*)d_in[13];
    const float* W3  = (const float*)d_in[14];
    const float* a3s = (const float*)d_in[15];
    const float* a3d = (const float*)d_in[16];
    const float* b3  = (const float*)d_in[17];
    const float* lw3 = (const float*)d_in[18];
    const float* lb3 = (const float*)d_in[19];
    float* out = (float*)d_out;

    // workspace layout (256B-aligned chunks)
    char* ws = (char*)d_ws;
    size_t off = 0;
    auto take = [&](size_t bytes) -> void* {
        void* p = ws + off;
        off += (bytes + 255) & ~(size_t)255;
        return p;
    };
    float*    h     = (float*)   take((size_t)NN * 1536 * 4);
    float*    skip  = (float*)   take((size_t)NN * 1536 * 4);
    float*    agg   = (float*)   take((size_t)NN * 1536 * 4);
    float*    x1    = (float*)   take((size_t)NN * 1024 * 4);
    float*    x2    = (float*)   take((size_t)NN * 1024 * 4);
    float*    asb   = (float*)   take((size_t)NN * 6 * 4);
    float*    adb   = (float*)   take((size_t)NN * 6 * 4);
    unsigned* emax  = (unsigned*)take((size_t)NN * 6 * 4);
    float*    denom = (float*)   take((size_t)NN * 6 * 4);
    float*    coef  = (float*)   take((size_t)ETOT * 6 * 4);
    // bf16 staging buffers
    bf16*     xbf   = (bf16*)    take((size_t)NN * 32 * 2);     // x padded 14->32
    bf16*     w1bf  = (bf16*)    take((size_t)1024 * 32 * 2);
    bf16*     lw1bf = (bf16*)    take((size_t)1024 * 32 * 2);
    bf16*     w2bf  = (bf16*)    take((size_t)1024 * 1024 * 2);
    bf16*     lw2bf = (bf16*)    take((size_t)1024 * 1024 * 2);
    bf16*     w3bf  = (bf16*)    take((size_t)1536 * 1024 * 2);
    bf16*     lw3bf = (bf16*)    take((size_t)256  * 1024 * 2);
    bf16*     x1bf  = (bf16*)    take((size_t)NN * 1024 * 2);
    bf16*     x2bf  = (bf16*)    take((size_t)NN * 1024 * 2);
    bf16*     hbf   = (bf16*)    take((size_t)NN * 1536 * 2);

    // one-shot bf16 conversions (K padded to multiples of 32)
    cvt(x,   xbf,   NN,   14,   32,   stream);
    cvt(W1,  w1bf,  1024, 14,   32,   stream);
    cvt(lw1, lw1bf, 1024, 14,   32,   stream);
    cvt(W2,  w2bf,  1024, 1024, 1024, stream);
    cvt(lw2, lw2bf, 1024, 1024, 1024, stream);
    cvt(W3,  w3bf,  1536, 1024, 1024, stream);
    cvt(lw3, lw3bf, 256,  1024, 1024, stream);

    // layer 1: 14(->32) -> 4 heads x 256, concat + ELU
    gat_layer<4, true>(xbf, 32, w1bf, lw1bf, a1s, a1d, b1, lb1,
                       h, hbf, skip, agg, asb, adb, emax, denom, coef, ei, x1, stream);
    cvt(x1, x1bf, NN, 1024, 1024, stream);
    // layer 2: 1024 -> 4 heads x 256, concat + ELU
    gat_layer<4, true>(x1bf, 1024, w2bf, lw2bf, a2s, a2d, b2, lb2,
                       h, hbf, skip, agg, asb, adb, emax, denom, coef, ei, x2, stream);
    cvt(x2, x2bf, NN, 1024, 1024, stream);
    // layer 3: 1024 -> 6 heads x 256, mean over heads, no ELU
    gat_layer<6, false>(x2bf, 1024, w3bf, lw3bf, a3s, a3d, b3, lb3,
                        h, hbf, skip, agg, asb, adb, emax, denom, coef, ei, out, stream);
}
